// RWKV7Block_22179211116822
// MI455X (gfx1250) — matrix operand order repaired and checked by reading, not compiled, not run on hardware
//
#include <hip/hip_runtime.h>
#include <cstdint>

#define B_    8
#define T_    1024
#define DIM_  1024
#define H_    16
#define HEAD_ 64
#define M_    (B_ * T_)   // 8192 token rows

typedef __bf16 bf16;
typedef bf16  v16bf __attribute__((ext_vector_type(16)));
typedef float v8f   __attribute__((ext_vector_type(8)));

#ifndef __has_builtin
#define __has_builtin(x) 0
#endif
#if __has_builtin(__builtin_amdgcn_global_load_async_to_lds_b128)
#define HAVE_ASYNC_LDS 1
#else
#define HAVE_ASYNC_LDS 0
#endif

__device__ __forceinline__ unsigned short f2bf(float x) {
  union { float f; unsigned int u; } c; c.f = x;
  unsigned int r = c.u + 0x7FFFu + ((c.u >> 16) & 1u);  // round-to-nearest-even
  return (unsigned short)(r >> 16);
}
__device__ __forceinline__ float sigm(float x) { return 1.f / (1.f + expf(-x)); }

#if HAVE_ASYNC_LDS
typedef int vi4 __attribute__((vector_size(16)));
typedef __attribute__((address_space(1))) vi4* gvi4p;
typedef __attribute__((address_space(3))) vi4* lvi4p;
__device__ __forceinline__ void async_cp16(const unsigned short* g, unsigned short* l) {
  __builtin_amdgcn_global_load_async_to_lds_b128(
      (gvi4p)(void*)g, (lvi4p)(void*)l, 0, 0);
}
#endif
__device__ __forceinline__ void wait_async() {
#if HAVE_ASYNC_LDS
#if __has_builtin(__builtin_amdgcn_s_wait_asynccnt)
  __builtin_amdgcn_s_wait_asynccnt(0);
#else
  asm volatile("s_wait_asynccnt 0" ::: "memory");
#endif
#endif
}

enum { ACT_NONE = 0, ACT_RELU = 1, ACT_TANH = 2, ACT_SIGMOID = 3 };

// ---------------------------------------------------------------------------
// bf16 WMMA GEMM: C[M,N] = A[M,K] * Bt[N,K]^T   (Bt = pre-transposed weights,
// N padded to 128 so tile loads need no guards).
//  A source: either (xn f32 + token-shift mix, fused) or direct bf16.
//  LDS double-buffered; tiles staged with GLOBAL_LOAD_ASYNC_TO_LDS_B128 when
//  available; epilogue: optional bias, activation, residual, f32/bf16 store.
// ---------------------------------------------------------------------------
#define TM 128
#define TN 128
#define TK 32
#define LDSK 40   // padded K-stride (ushort units)

__global__ __launch_bounds__(256)
void gemm_wmma(const float* __restrict__ xn, const float* __restrict__ mix,
               const unsigned short* __restrict__ Abf,
               const unsigned short* __restrict__ Bt,
               float* __restrict__ Cf, unsigned short* __restrict__ Cb,
               const float* __restrict__ bias, const float* __restrict__ res,
               int M, int N, int K, int act)
{
  __shared__ __align__(16) unsigned short As[2 * TM * LDSK];
  __shared__ __align__(16) unsigned short Bs[2 * TN * LDSK];

  const int tid  = threadIdx.x;
  const int lane = tid & 31;
  const int wave = tid >> 5;
  const int wm = wave >> 1, wn = wave & 1;       // 4x2 wave grid
  const int m0 = blockIdx.y * TM, n0 = blockIdx.x * TN;
  const int wrow = wm * 32, wcol = wn * 64;
  const int l16 = lane & 15, lh = lane >> 4;

  union AccU { v8f v; float f[8]; } acc[2][4];
  #pragma unroll
  for (int i = 0; i < 2; ++i)
    #pragma unroll
    for (int j = 0; j < 4; ++j)
      #pragma unroll
      for (int e = 0; e < 8; ++e) acc[i][j].f[e] = 0.f;

  // stage tile kk into LDS buffer sbuf
  auto stage = [&](int sbuf, int kk) {
    unsigned short* Asb = As + sbuf * (TM * LDSK);
    unsigned short* Bsb = Bs + sbuf * (TN * LDSK);
    if (xn != nullptr) {
      // fused token-shift mix: a = cur + (prev - cur) * mix[k], prev=0 at t==0
      #pragma unroll
      for (int it = 0; it < 4; ++it) {
        int idx = (it * 256 + tid) * 4;
        int row = idx >> 5, kc = idx & 31;
        long grow = (long)(m0 + row);
        const float* cp = xn + grow * (long)K + kk + kc;
        int tmod = (int)grow & (T_ - 1);
        float sc = tmod ? 1.f : 0.f;
        const float* pp = cp - (tmod ? K : 0);
        float4 cur = *(const float4*)cp;
        float4 prv = *(const float4*)pp;
        float4 mx  = *(const float4*)(mix + kk + kc);
        unsigned int u0 =
            (unsigned int)f2bf(cur.x + (prv.x * sc - cur.x) * mx.x)
          | ((unsigned int)f2bf(cur.y + (prv.y * sc - cur.y) * mx.y) << 16);
        unsigned int u1 =
            (unsigned int)f2bf(cur.z + (prv.z * sc - cur.z) * mx.z)
          | ((unsigned int)f2bf(cur.w + (prv.w * sc - cur.w) * mx.w) << 16);
        *(uint2*)&Asb[row * LDSK + kc] = make_uint2(u0, u1);
      }
    } else {
#if HAVE_ASYNC_LDS
      #pragma unroll
      for (int it = 0; it < 2; ++it) {
        int task = it * 256 + tid;
        int row = task >> 2, ch = (task & 3) * 8;
        async_cp16(Abf + (long)(m0 + row) * K + kk + ch, &Asb[row * LDSK + ch]);
      }
#else
      #pragma unroll
      for (int it = 0; it < 4; ++it) {
        int idx = (it * 256 + tid) * 4;
        int row = idx >> 5, kc = idx & 31;
        *(uint2*)&Asb[row * LDSK + kc] =
            *(const uint2*)(Abf + (long)(m0 + row) * K + kk + kc);
      }
#endif
    }
#if HAVE_ASYNC_LDS
    #pragma unroll
    for (int it = 0; it < 2; ++it) {
      int task = it * 256 + tid;
      int row = task >> 2, ch = (task & 3) * 8;
      async_cp16(Bt + (long)(n0 + row) * K + kk + ch, &Bsb[row * LDSK + ch]);
    }
#else
    #pragma unroll
    for (int it = 0; it < 4; ++it) {
      int idx = (it * 256 + tid) * 4;
      int row = idx >> 5, kc = idx & 31;
      *(uint2*)&Bsb[row * LDSK + kc] =
          *(const uint2*)(Bt + (long)(n0 + row) * K + kk + kc);
    }
#endif
  };

  stage(0, 0);
  const int nk = K / TK;
  for (int kt = 0; kt < nk; ++kt) {
    const int cur = kt & 1;
    wait_async();        // this wave's async copies for tile kt complete
    __syncthreads();     // all waves' tile kt visible; prev buffer reads done
    if (kt + 1 < nk) stage(cur ^ 1, (kt + 1) * TK);  // overlap with compute

    const unsigned short* Asb = As + cur * (TM * LDSK);
    const unsigned short* Bsb = Bs + cur * (TN * LDSK);
    union FragU { v16bf v; uint4 q[2]; } af[2], bfr[4];
    const int kbA = lh * 8;   // lanes 0-15: K0..7 & K16..23 ; 16-31: K8..15 & K24..31
    #pragma unroll
    for (int mi = 0; mi < 2; ++mi) {
      const unsigned short* p = &Asb[(wrow + mi * 16 + l16) * LDSK + kbA];
      af[mi].q[0] = *(const uint4*)p;
      af[mi].q[1] = *(const uint4*)(p + 16);
    }
    const int kbB = lh * 8;   // lanes 0-15: K0..7 & K16..23 ; 16-31: K8..15 & K24..31 (col = lane&15)
    #pragma unroll
    for (int ni = 0; ni < 4; ++ni) {
      const unsigned short* p = &Bsb[(wcol + ni * 16 + l16) * LDSK + kbB];
      bfr[ni].q[0] = *(const uint4*)p;
      bfr[ni].q[1] = *(const uint4*)(p + 16);
    }
    #pragma unroll
    for (int mi = 0; mi < 2; ++mi)
      #pragma unroll
      for (int ni = 0; ni < 4; ++ni)
        acc[mi][ni].v = __builtin_amdgcn_wmma_f32_16x16x32_bf16(
            false, af[mi].v, false, bfr[ni].v, (short)0, acc[mi][ni].v, false, false);
  }

  // ---- epilogue: C layout VGPR e -> M = e + 8*(lane>=16), N = lane&15 ----
  #pragma unroll
  for (int ni = 0; ni < 4; ++ni) {
    int col = n0 + wcol + ni * 16 + l16;
    if (col >= N) continue;
    float bv = bias ? bias[col] : 0.f;
    #pragma unroll
    for (int mi = 0; mi < 2; ++mi) {
      #pragma unroll
      for (int e = 0; e < 8; ++e) {
        int row = m0 + wrow + mi * 16 + e + (lh << 3);
        float val = acc[mi][ni].f[e] + bv;
        if      (act == ACT_RELU)    val = fmaxf(val, 0.f);
        else if (act == ACT_TANH)    val = tanhf(val);
        else if (act == ACT_SIGMOID) val = sigm(val);
        long o = (long)row * N + col;
        if (res) val += res[o];
        if (Cf) Cf[o] = val; else Cb[o] = f2bf(val);
      }
    }
  }
}

// ---------------------------------------------------------------------------
// Weight convert+transpose: src f32 [K,N] -> dst bf16 [Npad,K] (rows>=N zero).
__global__ __launch_bounds__(256)
void cvt_bf16_t(const float* __restrict__ src, unsigned short* __restrict__ dst,
                int K, int N, int Npad) {
  __shared__ float tile[32][33];
  const int kb = blockIdx.y * 32, nb = blockIdx.x * 32;
  const int tx = threadIdx.x & 31, ty = threadIdx.x >> 5;  // 32 x 8
  #pragma unroll
  for (int i = 0; i < 32; i += 8) {
    int k = kb + ty + i, n = nb + tx;
    tile[ty + i][tx] = (n < N) ? src[(long)k * N + n] : 0.f;
  }
  __syncthreads();
  #pragma unroll
  for (int i = 0; i < 32; i += 8) {
    int n = nb + ty + i, k = kb + tx;
    if (n < Npad) dst[(long)n * K + k] = f2bf(tile[tx][ty + i]);
  }
}

// LayerNorm over rows of 1024; f32 or bf16 output.
__global__ __launch_bounds__(256)
void layernorm_k(const float* __restrict__ x, const float* __restrict__ w,
                 const float* __restrict__ b, float* __restrict__ of,
                 unsigned short* __restrict__ ob) {
  const int row = blockIdx.x, tid = threadIdx.x;
  __shared__ float s1[256], s2[256];
  const float* xr = x + (long)row * DIM_;
  float a0 = xr[tid], a1 = xr[tid + 256], a2 = xr[tid + 512], a3 = xr[tid + 768];
  s1[tid] = a0 + a1 + a2 + a3;
  s2[tid] = a0 * a0 + a1 * a1 + a2 * a2 + a3 * a3;
  __syncthreads();
  for (int off = 128; off > 0; off >>= 1) {
    if (tid < off) { s1[tid] += s1[tid + off]; s2[tid] += s2[tid + off]; }
    __syncthreads();
  }
  float mu = s1[0] * (1.f / DIM_);
  float var = s2[0] * (1.f / DIM_) - mu * mu;
  float rs = rsqrtf(var + 1e-5f);
  #pragma unroll
  for (int i = 0; i < 4; ++i) {
    int c = tid + i * 256;
    float v = (xr[c] - mu) * rs * w[c] + b[c];
    long o = (long)row * DIM_ + c;
    if (of) of[o] = v; else ob[o] = f2bf(v);
  }
}

// Elementwise prep of scan inputs; k/v/w transformed in place, aa/bb written.
__global__ __launch_bounds__(128)
void prep_scan(float* __restrict__ kbuf, float* __restrict__ vbuf,
               const float* __restrict__ vres, const float* __restrict__ vfirst,
               float* __restrict__ wbuf, const float* __restrict__ apre,
               float* __restrict__ aabuf, float* __restrict__ bbbuf,
               const float* __restrict__ k_k, const float* __restrict__ k_a,
               const float* __restrict__ w0, const float* __restrict__ a0,
               const float* __restrict__ v0) {
  const long row = blockIdx.x;
  const int tid = threadIdx.x, head = tid >> 3;
  __shared__ float part[128];
  float kk8[8], av8[8];
  const long base = row * DIM_ + tid * 8;
  float ps = 0.f;
  #pragma unroll
  for (int e = 0; e < 8; ++e) {
    int c = tid * 8 + e; long idx = base + e;
    float kr = kbuf[idx];
    float kkv = kr * k_k[c];
    kk8[e] = kkv; ps += kkv * kkv;
    float av = sigm(a0[c] + apre[idx]);
    av8[e] = av;
    kbuf[idx] = kr * (1.f + (av - 1.f) * k_a[c]);
    float wl = w0[c] + wbuf[idx];
    float w = -log1pf(expf(-wl)) - 0.5f;     // -softplus(-y) - 0.5
    wbuf[idx] = expf(-expf(w));              // decay in (0,1)
    float vr = vbuf[idx];
    vbuf[idx] = vr + (vfirst[idx] - vr) * sigm(v0[c] + vres[idx]);
  }
  part[tid] = ps;
  __syncthreads();
  float hs = 0.f;
  #pragma unroll
  for (int i = 0; i < 8; ++i) hs += part[head * 8 + i];
  float rn = 1.f / fmaxf(sqrtf(hs), 1e-12f);
  #pragma unroll
  for (int e = 0; e < 8; ++e) {
    long idx = base + e;
    float kkn = kk8[e] * rn;
    aabuf[idx] = -kkn;
    bbbuf[idx] = kkn * av8[e];
  }
}

// Sequential RWKV-7 scan: one block (64 threads) per (b,h); thread owns state row.
#define CH 8
__global__ __launch_bounds__(64)
void rwkv_scan(const float* __restrict__ wd, const float* __restrict__ aa,
               const float* __restrict__ bb, const float* __restrict__ kk,
               const float* __restrict__ rr, const float* __restrict__ vv,
               float* __restrict__ out) {
  const int bh = blockIdx.x, b = bh / H_, h = bh % H_;
  const int tid = threadIdx.x;
  const long base = (long)b * T_ * DIM_ + h * HEAD_ + tid;  // channel = h*64+j
  __shared__ __align__(16) float sw[CH][HEAD_], sa[CH][HEAD_], sb[CH][HEAD_],
                                 sk[CH][HEAD_], sr[CH][HEAD_], sv[CH][HEAD_];
  float s[HEAD_];
  #pragma unroll
  for (int j = 0; j < HEAD_; ++j) s[j] = 0.f;

  for (int t0 = 0; t0 < T_; t0 += CH) {
    #pragma unroll
    for (int c = 0; c < CH; ++c) {
      long off = base + (long)(t0 + c) * DIM_;
      sw[c][tid] = wd[off]; sa[c][tid] = aa[off]; sb[c][tid] = bb[off];
      sk[c][tid] = kk[off]; sr[c][tid] = rr[off]; sv[c][tid] = vv[off];
    }
    __syncthreads();
    #pragma unroll 1
    for (int c = 0; c < CH; ++c) {
      const float4* A4 = (const float4*)sa[c];
      const float4* B4 = (const float4*)sb[c];
      const float4* K4 = (const float4*)sk[c];
      const float4* W4 = (const float4*)sw[c];
      const float4* R4 = (const float4*)sr[c];
      float sacc = 0.f;
      #pragma unroll
      for (int j = 0; j < 16; ++j) {
        float4 av = A4[j];
        sacc = fmaf(s[4*j+0], av.x, sacc); sacc = fmaf(s[4*j+1], av.y, sacc);
        sacc = fmaf(s[4*j+2], av.z, sacc); sacc = fmaf(s[4*j+3], av.w, sacc);
      }
      float vi = sv[c][tid], oacc = 0.f;
      #pragma unroll
      for (int j = 0; j < 16; ++j) {
        float4 bv = B4[j], kv = K4[j], wv = W4[j], rv = R4[j];
        s[4*j+0] = fmaf(s[4*j+0], wv.x, fmaf(sacc, bv.x, vi * kv.x));
        oacc = fmaf(s[4*j+0], rv.x, oacc);
        s[4*j+1] = fmaf(s[4*j+1], wv.y, fmaf(sacc, bv.y, vi * kv.y));
        oacc = fmaf(s[4*j+1], rv.y, oacc);
        s[4*j+2] = fmaf(s[4*j+2], wv.z, fmaf(sacc, bv.z, vi * kv.z));
        oacc = fmaf(s[4*j+2], rv.z, oacc);
        s[4*j+3] = fmaf(s[4*j+3], wv.w, fmaf(sacc, bv.w, vi * kv.w));
        oacc = fmaf(s[4*j+3], rv.w, oacc);
      }
      out[base + (long)(t0 + c) * DIM_] = oacc;
    }
    __syncthreads();
  }
}

// Per-head GroupNorm + bonus + gate -> bf16 (o*g) for Wo GEMM.
__global__ __launch_bounds__(128)
void post_gn(const float* __restrict__ o, const float* __restrict__ r,
             const float* __restrict__ k, const float* __restrict__ v,
             const float* __restrict__ g, const float* __restrict__ r_k,
             const float* __restrict__ lnw, const float* __restrict__ lnb,
             unsigned short* __restrict__ og) {
  const long row = blockIdx.x;
  const int tid = threadIdx.x, head = tid >> 3;
  __shared__ float pm[128], pq[128], pb[128];
  const long base = row * DIM_ + tid * 8;
  float o8[8], sm = 0.f, sq = 0.f, sbn = 0.f;
  #pragma unroll
  for (int e = 0; e < 8; ++e) {
    int c = tid * 8 + e; long idx = base + e;
    float ov = o[idx]; o8[e] = ov;
    sm += ov; sq += ov * ov;
    sbn += r[idx] * k[idx] * r_k[c];
  }
  pm[tid] = sm; pq[tid] = sq; pb[tid] = sbn;
  __syncthreads();
  float m = 0.f, q = 0.f, bn = 0.f;
  #pragma unroll
  for (int i = 0; i < 8; ++i) {
    m += pm[head * 8 + i]; q += pq[head * 8 + i]; bn += pb[head * 8 + i];
  }
  m *= (1.f / HEAD_);
  float var = q * (1.f / HEAD_) - m * m;
  float rs = rsqrtf(var + 64e-5f);
  #pragma unroll
  for (int e = 0; e < 8; ++e) {
    int c = tid * 8 + e; long idx = base + e;
    float val = (o8[e] - m) * rs * lnw[c] + lnb[c];
    val += bn * v[idx];
    og[idx] = f2bf(val * g[idx]);
  }
}

// ---------------------------------------------------------------------------
extern "C" void kernel_launch(void* const* d_in, const int* in_sizes, int n_in,
                              void* d_out, int out_size, void* d_ws, size_t ws_size,
                              hipStream_t stream) {
  (void)in_sizes; (void)n_in; (void)out_size; (void)ws_size;
  const float* x       = (const float*)d_in[0];
  const float* v_first = (const float*)d_in[1];
  const float* n1_w = (const float*)d_in[2];  const float* n1_b = (const float*)d_in[3];
  const float* n2_w = (const float*)d_in[4];  const float* n2_b = (const float*)d_in[5];
  const float* mw1  = (const float*)d_in[6];  const float* mb1  = (const float*)d_in[7];
  const float* mw2  = (const float*)d_in[8];  const float* mb2  = (const float*)d_in[9];
  const float* x_r = (const float*)d_in[10];  const float* x_w = (const float*)d_in[11];
  const float* x_k = (const float*)d_in[12];  const float* x_v = (const float*)d_in[13];
  const float* x_a = (const float*)d_in[14];  const float* x_g = (const float*)d_in[15];
  const float* Wr = (const float*)d_in[16];   const float* Wk = (const float*)d_in[17];
  const float* Wv = (const float*)d_in[18];   const float* Wo = (const float*)d_in[19];
  const float* w0 = (const float*)d_in[20];   const float* w1 = (const float*)d_in[21];
  const float* w2 = (const float*)d_in[22];   const float* a0 = (const float*)d_in[23];
  const float* a1 = (const float*)d_in[24];   const float* a2 = (const float*)d_in[25];
  const float* v0 = (const float*)d_in[26];   const float* v1 = (const float*)d_in[27];
  const float* v2 = (const float*)d_in[28];   const float* g1 = (const float*)d_in[29];
  const float* g2 = (const float*)d_in[30];
  const float* k_k = (const float*)d_in[31];  const float* k_a = (const float*)d_in[32];
  const float* r_k = (const float*)d_in[33];
  const float* ln_w = (const float*)d_in[34]; const float* ln_b = (const float*)d_in[35];
  float* out = (float*)d_out;

  char* ws = (char*)d_ws;
  size_t off = 0;
  auto take = [&](size_t bytes) -> void* {
    void* p = ws + off; off += bytes; off = (off + 255) & ~(size_t)255; return p;
  };
  const size_t MD = (size_t)M_ * DIM_;
  // bf16 transposed weights [Npad, K]
  unsigned short* WrT  = (unsigned short*)take(1024 * 1024 * 2);
  unsigned short* WkT  = (unsigned short*)take(1024 * 1024 * 2);
  unsigned short* WvT  = (unsigned short*)take(1024 * 1024 * 2);
  unsigned short* WoT  = (unsigned short*)take(1024 * 1024 * 2);
  unsigned short* w1T  = (unsigned short*)take(128 * 1024 * 2);
  unsigned short* w2T  = (unsigned short*)take(1024 * 64 * 2);
  unsigned short* a1T  = (unsigned short*)take(128 * 1024 * 2);
  unsigned short* a2T  = (unsigned short*)take(1024 * 64 * 2);
  unsigned short* v1T  = (unsigned short*)take(128 * 1024 * 2);
  unsigned short* v2T  = (unsigned short*)take(1024 * 32 * 2);
  unsigned short* g1T  = (unsigned short*)take(128 * 1024 * 2);
  unsigned short* g2T  = (unsigned short*)take(1024 * 128 * 2);
  unsigned short* mw1T = (unsigned short*)take(2048 * 1024 * 2);
  unsigned short* mw2T = (unsigned short*)take(1024 * 2048 * 2);
  // f32 activations
  float* xn    = (float*)take(MD * 4);
  float* rbuf  = (float*)take(MD * 4);
  float* kbuf  = (float*)take(MD * 4);
  float* vbuf  = (float*)take(MD * 4);
  float* wbuf  = (float*)take(MD * 4);   // wlog -> decay in place
  float* apre  = (float*)take(MD * 4);
  float* vres  = (float*)take(MD * 4);
  float* gbuf  = (float*)take(MD * 4);
  float* aabuf = (float*)take(MD * 4);
  float* bbbuf = (float*)take(MD * 4);
  float* obuf  = (float*)take(MD * 4);
  float* x1    = (float*)take(MD * 4);
  // bf16 activations
  unsigned short* hw_b = (unsigned short*)take((size_t)M_ * 64 * 2);
  unsigned short* ha_b = (unsigned short*)take((size_t)M_ * 64 * 2);
  unsigned short* hv_b = (unsigned short*)take((size_t)M_ * 32 * 2);
  unsigned short* hg_b = (unsigned short*)take((size_t)M_ * 128 * 2);
  unsigned short* og_b = (unsigned short*)take(MD * 2);
  unsigned short* h2_b = (unsigned short*)take(MD * 2);
  unsigned short* m1_b = (unsigned short*)take((size_t)M_ * 2048 * 2);

  auto cvtT = [&](const float* s, unsigned short* d, int K, int N, int Npad) {
    dim3 grid(Npad / 32, K / 32);
    cvt_bf16_t<<<grid, 256, 0, stream>>>(s, d, K, N, Npad);
  };
  cvtT(Wr, WrT, 1024, 1024, 1024);  cvtT(Wk, WkT, 1024, 1024, 1024);
  cvtT(Wv, WvT, 1024, 1024, 1024);  cvtT(Wo, WoT, 1024, 1024, 1024);
  cvtT(w1, w1T, 1024, 64, 128);     cvtT(w2, w2T, 64, 1024, 1024);
  cvtT(a1, a1T, 1024, 64, 128);     cvtT(a2, a2T, 64, 1024, 1024);
  cvtT(v1, v1T, 1024, 32, 128);     cvtT(v2, v2T, 32, 1024, 1024);
  cvtT(g1, g1T, 1024, 128, 128);    cvtT(g2, g2T, 128, 1024, 1024);
  cvtT(mw1, mw1T, 1024, 2048, 2048); cvtT(mw2, mw2T, 2048, 1024, 1024);

  // LN1
  layernorm_k<<<M_, 256, 0, stream>>>(x, n1_w, n1_b, xn, nullptr);

  auto gemm = [&](const float* xns, const float* mix, const unsigned short* A,
                  const unsigned short* BtM, float* Cf, unsigned short* Cb,
                  const float* bias, const float* res, int N, int K, int act) {
    dim3 grid((N + TN - 1) / TN, M_ / TM);
    gemm_wmma<<<grid, 256, 0, stream>>>(xns, mix, A, BtM, Cf, Cb, bias, res,
                                        M_, N, K, act);
  };

  // main projections (token-shift fused into A load)
  gemm(xn, x_r, nullptr, WrT, rbuf, nullptr, nullptr, nullptr, DIM_, DIM_, ACT_NONE);
  gemm(xn, x_k, nullptr, WkT, kbuf, nullptr, nullptr, nullptr, DIM_, DIM_, ACT_NONE);
  gemm(xn, x_v, nullptr, WvT, vbuf, nullptr, nullptr, nullptr, DIM_, DIM_, ACT_NONE);
  // LoRA down-projections
  gemm(xn, x_w, nullptr, w1T, nullptr, hw_b, nullptr, nullptr, 64, DIM_, ACT_TANH);
  gemm(xn, x_a, nullptr, a1T, nullptr, ha_b, nullptr, nullptr, 64, DIM_, ACT_NONE);
  gemm(xn, x_v, nullptr, v1T, nullptr, hv_b, nullptr, nullptr, 32, DIM_, ACT_NONE);
  gemm(xn, x_g, nullptr, g1T, nullptr, hg_b, nullptr, nullptr, 128, DIM_, ACT_SIGMOID);
  // LoRA up-projections
  gemm(nullptr, nullptr, hw_b, w2T, wbuf, nullptr, nullptr, nullptr, DIM_, 64, ACT_NONE);
  gemm(nullptr, nullptr, ha_b, a2T, apre, nullptr, nullptr, nullptr, DIM_, 64, ACT_NONE);
  gemm(nullptr, nullptr, hv_b, v2T, vres, nullptr, nullptr, nullptr, DIM_, 32, ACT_NONE);
  gemm(nullptr, nullptr, hg_b, g2T, gbuf, nullptr, nullptr, nullptr, DIM_, 128, ACT_NONE);

  // scan prep + sequential scan + groupnorm/gate
  prep_scan<<<M_, 128, 0, stream>>>(kbuf, vbuf, vres, v_first, wbuf, apre,
                                    aabuf, bbbuf, k_k, k_a, w0, a0, v0);
  rwkv_scan<<<B_ * H_, 64, 0, stream>>>(wbuf, aabuf, bbbuf, kbuf, rbuf, vbuf, obuf);
  post_gn<<<M_, 128, 0, stream>>>(obuf, rbuf, kbuf, vbuf, gbuf, r_k, ln_w, ln_b, og_b);

  // output projection with fused residual: x1 = x + og @ Wo
  gemm(nullptr, nullptr, og_b, WoT, x1, nullptr, nullptr, x, DIM_, DIM_, ACT_NONE);

  // MLP: out = x1 + relu(LN2(x1) @ mw1 + b1) @ mw2 + b2  (residual fused)
  layernorm_k<<<M_, 256, 0, stream>>>(x1, n2_w, n2_b, nullptr, h2_b);
  gemm(nullptr, nullptr, h2_b, mw1T, nullptr, m1_b, mb1, nullptr, 2048, DIM_, ACT_RELU);
  gemm(nullptr, nullptr, m1_b, mw2T, out, nullptr, mb2, x1, DIM_, 2048, ACT_NONE);

  // second output: v_first passthrough
  (void)hipMemcpyAsync(out + MD, v_first, MD * sizeof(float),
                       hipMemcpyDeviceToDevice, stream);
}
